// TFJSP_Encoder_GTrans_21414706937974
// MI455X (gfx1250) — compile-verified, matrix operand order checked by reading, not run on hardware
//
#include <hip/hip_runtime.h>
#include <hip/hip_bf16.h>

// ---------------------------------------------------------------------------
// TFJSP graph-transformer encoder forward for MI455X (gfx1250, wave32, WMMA).
// All GEMMs run on v_wmma_f32_16x16x32_bf16 (bf16 in, f32 accumulate).
// Fragments are assembled from contiguous 16-byte runs (ds_load_b128 /
// global_load_b128), not scalar u16 loads.
// ---------------------------------------------------------------------------

typedef __attribute__((ext_vector_type(16))) __bf16 v16bf;
typedef __attribute__((ext_vector_type(8)))  __bf16 v8bf;
typedef __attribute__((ext_vector_type(4)))  __bf16 v4bf;
typedef __attribute__((ext_vector_type(8)))  float  v8f;
typedef __attribute__((ext_vector_type(4)))  float  v4f;

static __device__ __forceinline__ v8f wmma_bf16(v16bf a, v16bf b, v8f c) {
  return __builtin_amdgcn_wmma_f32_16x16x32_bf16(false, a, false, b,
                                                 (short)0, c, false, false);
}
static __device__ __forceinline__ __bf16 bfc(float x) { return (__bf16)x; }

static __device__ __forceinline__ v16bf cat16(v8bf lo, v8bf hi) {
  return __builtin_shufflevector(lo, hi, 0,1,2,3,4,5,6,7,8,9,10,11,12,13,14,15);
}
static __device__ __forceinline__ v4bf cvt4(v4f x) {
  v4bf r; r[0] = bfc(x[0]); r[1] = bfc(x[1]); r[2] = bfc(x[2]); r[3] = bfc(x[3]);
  return r;
}
static __device__ __forceinline__ v8bf cvt8(v4f x, v4f y) {
  v8bf r;
  r[0] = bfc(x[0]); r[1] = bfc(x[1]); r[2] = bfc(x[2]); r[3] = bfc(x[3]);
  r[4] = bfc(y[0]); r[5] = bfc(y[1]); r[6] = bfc(y[2]); r[7] = bfc(y[3]);
  return r;
}

// Fragment striping (documented 16-bit A(16x32) layout, B as its K<->lane
// mirror): one lane's fragment = two contiguous 8-element K runs at
// K = half*8 and K = 16 + half*8.

// ======================= WMMA GEMM: Y = act(X @ W + b) ======================
// X:[M,K] f32, W:[K,N] f32 (shared over batch), Y:[M,N] f32.
// M % 32 == 0, N % 128 == 0, K % 32 == 0 (holds for every call here).
#define GBM 32
#define GBN 128
#define GBK 32
#define KPAD 8   // row stride 40 bf16 = 80 B (16-B multiple)

__global__ __launch_bounds__(256) void wmma_gemm_kernel(
    const float* __restrict__ X, const float* __restrict__ W,
    const float* __restrict__ bias, float* __restrict__ Y,
    int M, int N, int K, int relu)
{
  __shared__ alignas(16) __bf16 As [GBM][GBK + KPAD];   // [m][k]
  __shared__ alignas(16) __bf16 BsT[GBN][GBK + KPAD];   // [n][k] (transposed)

  const int m0   = blockIdx.x * GBM;
  const int n0   = blockIdx.y * GBN;
  const int tid  = threadIdx.x;
  const int wave = tid >> 5;
  const int lane = tid & 31;
  const int wm   = wave >> 2;   // 0..1  (M sub-tile)
  const int wn   = wave & 3;    // 0..3  (N sub-tile, 32 cols each)
  const int half = lane >> 4;
  const int l16  = lane & 15;

  v8f acc0 = {}; v8f acc1 = {};

  for (int k0 = 0; k0 < K; k0 += GBK) {
    // Stage A tile (32x32): one float4 per thread, packed v4bf LDS store.
    {
      int r  = tid >> 3;          // 0..31
      int c4 = (tid & 7) * 4;     // 0..28
      v4f xv = *(const v4f*)&X[(size_t)(m0 + r) * K + k0 + c4];
      *(v4bf*)&As[r][c4] = cvt4(xv);
    }
    // Stage B tile (32x128) transposed: float4 global reads, bf16 writes.
#pragma unroll
    for (int t = 0; t < 4; ++t) {
      int idx = tid + t * 256;    // 1024 quads
      int r   = idx >> 5;         // k row 0..31
      int c4  = (idx & 31) * 4;   // n col 0..124
      v4f wv = *(const v4f*)&W[(size_t)(k0 + r) * N + n0 + c4];
      BsT[c4 + 0][r] = bfc(wv[0]);
      BsT[c4 + 1][r] = bfc(wv[1]);
      BsT[c4 + 2][r] = bfc(wv[2]);
      BsT[c4 + 3][r] = bfc(wv[3]);
    }
    __syncthreads();

    const __bf16* arow = &As[wm * 16 + l16][0];
    v16bf a = cat16(*(const v8bf*)(arow + half * 8),
                    *(const v8bf*)(arow + 16 + half * 8));
    const __bf16* b0r = &BsT[wn * 32 + l16][0];
    const __bf16* b1r = &BsT[wn * 32 + 16 + l16][0];
    v16bf b0 = cat16(*(const v8bf*)(b0r + half * 8),
                     *(const v8bf*)(b0r + 16 + half * 8));
    v16bf b1 = cat16(*(const v8bf*)(b1r + half * 8),
                     *(const v8bf*)(b1r + 16 + half * 8));

    acc0 = wmma_bf16(a, b0, acc0);
    acc1 = wmma_bf16(a, b1, acc1);
    __syncthreads();
  }

#pragma unroll
  for (int v = 0; v < 8; ++v) {
    int m  = m0 + wm * 16 + v + 8 * half;
    int na = n0 + wn * 32 + l16;
    int nb = na + 16;
    float r0 = acc0[v], r1 = acc1[v];
    if (bias) { r0 += bias[na]; r1 += bias[nb]; }
    if (relu) { r0 = fmaxf(r0, 0.f); r1 = fmaxf(r1, 0.f); }
    Y[(size_t)m * N + na] = r0;
    Y[(size_t)m * N + nb] = r1;
  }
}

// ============================ Fused attention ===============================
// One wave per (batch b, 16-row tile). Loops the 8 heads.
//   score = q.k'/4 + edge*we[h]  -> softmax -> o = attn @ v
//   Eout  = sum_h wout[h]*score + edge     (skipped when Eout == nullptr)
#define MAXNC 480

__global__ __launch_bounds__(32) void attn_kernel(
    const float* __restrict__ Q,    // [B,Nr,128] (head h at cols h*16..)
    const float* __restrict__ Km,   // [B,Nc,128]
    const float* __restrict__ Vm,   // [B,Nc,128]
    const float* __restrict__ edge, // [B,Nr,Nc]
    const float* __restrict__ we,   // [8]
    const float* __restrict__ wout, // [8]
    float* __restrict__ O,          // [B,Nr,128]
    float* __restrict__ Eout,       // [B,Nr,Nc] or nullptr
    int Nr, int Nc)
{
  __shared__ alignas(16) float  sc[16][MAXNC];     // score strip -> probs
  __shared__ alignas(16) float  ea[16][MAXNC];     // sum_h wout[h]*score
  __shared__ alignas(16) __bf16 Vt[16][32 + KPAD]; // transposed V chunk [d][j]

  const int b    = blockIdx.y;
  const int i0   = blockIdx.x * 16;
  const int lane = threadIdx.x;
  const int half = lane >> 4;
  const int l16  = lane & 15;
  const int ncT  = (Nc + 15) >> 4;
  const int Nc32 = (Nc + 31) & ~31;
  const float inv_s = 0.25f;        // 1/sqrt(DQ=16)

  for (int r = 0; r < 16; ++r)
    for (int j = lane; j < Nc; j += 32) ea[r][j] = 0.f;
  __syncthreads();

  for (int h = 0; h < 8; ++h) {
    // --- q A-fragment: two float4 loads of the head slice; K 16..31 = zeros.
    v16bf qa;
    {
      v8bf lo = {};
      int row = i0 + l16;
      if (row < Nr) {
        const float* qp = Q + ((size_t)b * Nr + row) * 128 + h * 16 + half * 8;
        lo = cvt8(*(const v4f*)qp, *(const v4f*)(qp + 4));
      }
      v8bf hz = {};
      qa = cat16(lo, hz);
    }

    // --- scores: q @ k^T per 16-column tile.
    for (int jt = 0; jt < ncT; ++jt) {
      int j0 = jt * 16;
      v16bf kf;
      {
        v8bf lo = {};
        int col = j0 + l16;
        if (col < Nc) {
          const float* kp = Km + ((size_t)b * Nc + col) * 128 + h * 16 + half * 8;
          lo = cvt8(*(const v4f*)kp, *(const v4f*)(kp + 4));
        }
        v8bf hz = {};
        kf = cat16(lo, hz);
      }
      v8f c = {};
      c = wmma_bf16(qa, kf, c);
#pragma unroll
      for (int v = 0; v < 8; ++v) {
        int m = v + 8 * half;
        int j = j0 + l16;
        if (j < Nc) {
          int gi = i0 + m;
          float e = (gi < Nr) ? edge[((size_t)b * Nr + gi) * Nc + j] : 0.f;
          sc[m][j] = c[v] * inv_s + e * we[h];
        }
      }
    }
    __syncthreads();

    // --- softmax per row (2 lanes per row) + edge-out accumulation.
    {
      const float wh = wout[h];
      int r = l16;
      float mx = -1e30f;
      for (int j = half; j < Nc; j += 2) mx = fmaxf(mx, sc[r][j]);
      mx = fmaxf(mx, __shfl_xor(mx, 16, 32));
      float sum = 0.f;
      for (int j = half; j < Nc; j += 2) {
        float s = sc[r][j];
        ea[r][j] += wh * s;
        float e = __expf(s - mx);
        sc[r][j] = e;
        sum += e;
      }
      sum += __shfl_xor(sum, 16, 32);
      float inv = 1.f / sum;
      for (int j = half; j < Nc; j += 2) sc[r][j] *= inv;
      // Zero-pad probs to the next multiple of 32 so AV needs no guards.
      for (int j = Nc + half; j < Nc32; j += 2) sc[r][j] = 0.f;
    }
    __syncthreads();

    // --- o = attn @ v : stage V chunk transposed, then all-b128 fragments.
    v8f oc = {};
    for (int j0 = 0; j0 < Nc32; j0 += 32) {
      {
        int j = j0 + lane;
        if (j < Nc) {
          const float* vp = Vm + ((size_t)b * Nc + j) * 128 + h * 16;
          v4f a0 = *(const v4f*)vp;
          v4f a1 = *(const v4f*)(vp + 4);
          v4f a2 = *(const v4f*)(vp + 8);
          v4f a3 = *(const v4f*)(vp + 12);
#pragma unroll
          for (int d = 0; d < 4; ++d) {
            Vt[d +  0][lane] = bfc(a0[d]);
            Vt[d +  4][lane] = bfc(a1[d]);
            Vt[d +  8][lane] = bfc(a2[d]);
            Vt[d + 12][lane] = bfc(a3[d]);
          }
        } else {
#pragma unroll
          for (int d = 0; d < 16; ++d) Vt[d][lane] = bfc(0.f);
        }
      }
      __syncthreads();

      const float* srow = &sc[l16][j0];
      v16bf aa = cat16(cvt8(*(const v4f*)(srow + half * 8),
                            *(const v4f*)(srow + half * 8 + 4)),
                       cvt8(*(const v4f*)(srow + 16 + half * 8),
                            *(const v4f*)(srow + 16 + half * 8 + 4)));
      const __bf16* vrow = &Vt[l16][0];
      v16bf bb = cat16(*(const v8bf*)(vrow + half * 8),
                       *(const v8bf*)(vrow + 16 + half * 8));
      oc = wmma_bf16(aa, bb, oc);
      __syncthreads();
    }
#pragma unroll
    for (int v = 0; v < 8; ++v) {
      int m = i0 + v + 8 * half;
      if (m < Nr) O[((size_t)b * Nr + m) * 128 + h * 16 + l16] = oc[v];
    }
    __syncthreads();
  }

  if (Eout) {
    for (int r = 0; r < 16; ++r) {
      int gi = i0 + r;
      if (gi >= Nr) break;
      for (int j = lane; j < Nc; j += 32)
        Eout[((size_t)b * Nr + gi) * Nc + j] =
            ea[r][j] + edge[((size_t)b * Nr + gi) * Nc + j];
    }
  }
}

// ===================== LayerNorm over residual (D = 128) ====================
__global__ __launch_bounds__(128) void add_ln_kernel(
    const float* __restrict__ Xa, const float* __restrict__ Xb,
    float* __restrict__ Y)
{
  __shared__ float red[128];
  const int row = blockIdx.x;
  const int d   = threadIdx.x;
  float x = Xa[(size_t)row * 128 + d] + Xb[(size_t)row * 128 + d];
  red[d] = x; __syncthreads();
  for (int s = 64; s > 0; s >>= 1) { if (d < s) red[d] += red[d + s]; __syncthreads(); }
  float mean = red[0] / 128.f; __syncthreads();
  float c = x - mean;
  red[d] = c * c; __syncthreads();
  for (int s = 64; s > 0; s >>= 1) { if (d < s) red[d] += red[d + s]; __syncthreads(); }
  float var = red[0] / 128.f;
  Y[(size_t)row * 128 + d] = c * rsqrtf(var + 1e-5f);
}

// ============================ Index plumbing ================================
// B=32, D=128 hardcoded (fixed by the reference problem).
__global__ void concat2_kernel(float* __restrict__ dst,
                               const float* __restrict__ s1,
                               const float* __restrict__ s2, int N1, int N2) {
  int total = 32 * (N1 + N2) * 128;
  int idx = blockIdx.x * 256 + threadIdx.x;
  if (idx >= total) return;
  int d = idx & 127;
  int r = (idx >> 7) % (N1 + N2);
  int b = idx / ((N1 + N2) << 7);
  dst[idx] = (r < N1) ? s1[((size_t)b * N1 + r) * 128 + d]
                      : s2[((size_t)b * N2 + (r - N1)) * 128 + d];
}

__global__ void concat3_kernel(float* __restrict__ dst,
                               const float* __restrict__ s1,
                               const float* __restrict__ s2,
                               const float* __restrict__ s3,
                               int N1, int N2, int N3) {
  int Nt = N1 + N2 + N3;
  int total = 32 * Nt * 128;
  int idx = blockIdx.x * 256 + threadIdx.x;
  if (idx >= total) return;
  int d = idx & 127;
  int r = (idx >> 7) % Nt;
  int b = idx / (Nt << 7);
  float v;
  if (r < N1)            v = s1[((size_t)b * N1 + r) * 128 + d];
  else if (r < N1 + N2)  v = s2[((size_t)b * N2 + (r - N1)) * 128 + d];
  else                   v = s3[((size_t)b * N3 + (r - N1 - N2)) * 128 + d];
  dst[idx] = v;
}

// edgeA[b,i,j] = j<50 ? proc[b,i,j] : off_OV[b,j-50,i]   (i<400, j<80)
__global__ void build_edgeA_kernel(float* __restrict__ dst,
                                   const float* __restrict__ proc,
                                   const float* __restrict__ offov) {
  int total = 32 * 400 * 80;
  int idx = blockIdx.x * 256 + threadIdx.x;
  if (idx >= total) return;
  int j = idx % 80; int i = (idx / 80) % 400; int b = idx / (80 * 400);
  dst[idx] = (j < 50) ? proc[((size_t)b * 400 + i) * 50 + j]
                      : offov[((size_t)b * 30 + (j - 50)) * 400 + i];
}

// edgeB[b,i,j] = j<50 ? onload[b,i,j] : proc[b,j-50,i]   (i<50, j<450)
__global__ void build_edgeB_kernel(float* __restrict__ dst,
                                   const float* __restrict__ onload,
                                   const float* __restrict__ proc) {
  int total = 32 * 50 * 450;
  int idx = blockIdx.x * 256 + threadIdx.x;
  if (idx >= total) return;
  int j = idx % 450; int i = (idx / 450) % 50; int b = idx / (450 * 50);
  dst[idx] = (j < 50) ? onload[((size_t)b * 50 + i) * 50 + j]
                      : proc[((size_t)b * 400 + (j - 50)) * 50 + i];
}

// proc' = po[:,:,:50] + op[:,:,50:]^T
__global__ void update_proc_kernel(float* __restrict__ pn,
                                   const float* __restrict__ po,
                                   const float* __restrict__ op) {
  int total = 32 * 400 * 50;
  int idx = blockIdx.x * 256 + threadIdx.x;
  if (idx >= total) return;
  int j = idx % 50; int i = (idx / 50) % 400; int b = idx / (50 * 400);
  pn[idx] = po[((size_t)b * 400 + i) * 80 + j]
          + op[((size_t)b * 50 + j) * 450 + 50 + i];
}

// off_OV' = po[:,:,50:]^T + eoC
__global__ void update_offov_kernel(float* __restrict__ on,
                                    const float* __restrict__ po,
                                    const float* __restrict__ eoC) {
  int total = 32 * 30 * 400;
  int idx = blockIdx.x * 256 + threadIdx.x;
  if (idx >= total) return;
  int i = idx % 400; int u = (idx / 400) % 30; int b = idx / (400 * 30);
  on[idx] = po[((size_t)b * 400 + i) * 80 + 50 + u] + eoC[idx];
}

// onload' = op[:,:,:50]
__global__ void update_onload_kernel(float* __restrict__ on,
                                     const float* __restrict__ op) {
  int total = 32 * 50 * 50;
  int idx = blockIdx.x * 256 + threadIdx.x;
  if (idx >= total) return;
  int j = idx % 50; int i = (idx / 50) % 50; int b = idx / (50 * 50);
  on[idx] = op[((size_t)b * 50 + i) * 450 + j];
}

// Assemble the 480x480 global edge matrix for the final block.
__global__ void build_ge_kernel(float* __restrict__ ge,
                                const float* __restrict__ proc,
                                const float* __restrict__ onload,
                                const float* __restrict__ offov) {
  int total = 32 * 480 * 480;
  int idx = blockIdx.x * 256 + threadIdx.x;
  if (idx >= total) return;
  int c = idx % 480; int r = (idx / 480) % 480; int b = idx / (480 * 480);
  float v = 0.f;
  if (r < 400 && c >= 400 && c < 450)            v = proc[((size_t)b*400 + r)*50 + (c-400)];
  else if (r >= 400 && r < 450 && c < 400)       v = proc[((size_t)b*400 + c)*50 + (r-400)];
  else if (r >= 400 && r < 450 && c >= 400 && c < 450)
                                                 v = onload[((size_t)b*50 + (r-400))*50 + (c-400)];
  else if (r >= 450 && c < 400)                  v = offov[((size_t)b*30 + (r-450))*400 + c];
  else if (r < 400 && c >= 450)                  v = offov[((size_t)b*30 + (c-450))*400 + r];
  ge[idx] = v;
}

// dst[B,Nr,128] = src[:, r0:r0+Nr, :]  (src rows = Nsrc)
__global__ void slice_kernel(float* __restrict__ dst, const float* __restrict__ src,
                             int r0, int Nr, int Nsrc) {
  int total = 32 * Nr * 128;
  int idx = blockIdx.x * 256 + threadIdx.x;
  if (idx >= total) return;
  int d = idx & 127;
  int r = (idx >> 7) % Nr;
  int b = idx / (Nr << 7);
  dst[idx] = src[((size_t)b * Nsrc + r0 + r) * 128 + d];
}

// ================================ Host side =================================
static inline int cdiv(int a, int b) { return (a + b - 1) / b; }

struct BlkW {
  const float *Wq, *Wk, *Wv, *Wo, *we, *wout, *W1, *b1, *W2, *b2;
};

static BlkW blk_weights(void* const* d_in, int i) {
  BlkW w;
  w.Wq   = (const float*)d_in[7]  + (size_t)i * 128 * 128;
  w.Wk   = (const float*)d_in[8]  + (size_t)i * 128 * 128;
  w.Wv   = (const float*)d_in[9]  + (size_t)i * 128 * 128;
  w.Wo   = (const float*)d_in[10] + (size_t)i * 128 * 128;
  w.we   = (const float*)d_in[11] + (size_t)i * 8;
  w.wout = (const float*)d_in[12] + (size_t)i * 8;
  w.W1   = (const float*)d_in[13] + (size_t)i * 128 * 512;
  w.b1   = (const float*)d_in[14] + (size_t)i * 512;
  w.W2   = (const float*)d_in[15] + (size_t)i * 512 * 128;
  w.b2   = (const float*)d_in[16] + (size_t)i * 128;
  return w;
}

static void run_block(const float* x_row, int Nr, const float* x_col, int Nc,
                      const float* edge, float* edge_out, const BlkW& w,
                      float* x_out, float* qb, float* kb, float* vb, float* ob,
                      float* pb, float* x1b, float* hb, hipStream_t stream) {
  const int Mr = 32 * Nr, Mc = 32 * Nc;
  wmma_gemm_kernel<<<dim3(Mr/32, 1), 256, 0, stream>>>(x_row, w.Wq, nullptr, qb, Mr, 128, 128, 0);
  wmma_gemm_kernel<<<dim3(Mc/32, 1), 256, 0, stream>>>(x_col, w.Wk, nullptr, kb, Mc, 128, 128, 0);
  wmma_gemm_kernel<<<dim3(Mc/32, 1), 256, 0, stream>>>(x_col, w.Wv, nullptr, vb, Mc, 128, 128, 0);
  attn_kernel<<<dim3(cdiv(Nr,16), 32), 32, 0, stream>>>(qb, kb, vb, edge, w.we, w.wout,
                                                        ob, edge_out, Nr, Nc);
  wmma_gemm_kernel<<<dim3(Mr/32, 1), 256, 0, stream>>>(ob, w.Wo, nullptr, pb, Mr, 128, 128, 0);
  add_ln_kernel<<<dim3(Mr), 128, 0, stream>>>(x_row, pb, x1b);
  wmma_gemm_kernel<<<dim3(Mr/32, 4), 256, 0, stream>>>(x1b, w.W1, w.b1, hb, Mr, 512, 128, 1);
  wmma_gemm_kernel<<<dim3(Mr/32, 1), 256, 0, stream>>>(hb,  w.W2, w.b2, pb, Mr, 128, 512, 0);
  add_ln_kernel<<<dim3(Mr), 128, 0, stream>>>(x1b, pb, x_out);
}

extern "C" void kernel_launch(void* const* d_in, const int* in_sizes, int n_in,
                              void* d_out, int out_size, void* d_ws, size_t ws_size,
                              hipStream_t stream) {
  (void)in_sizes; (void)n_in; (void)out_size; (void)ws_size;

  // ---- workspace bump allocator (f32 elements, 256B aligned) ----
  size_t off = 0;
  auto alloc = [&](size_t elems) {
    float* p = (float*)((char*)d_ws + off);
    off += elems * sizeof(float);
    off = (off + 255) & ~(size_t)255;
    return p;
  };
  const size_t S_OPE = 32*400*128, S_MA = 32*50*128, S_VEH = 32*30*128;
  const size_t S_PROC = 32*400*50, S_ONL = 32*50*50, S_OFF = 32*30*400;
  const size_t S_NODE = 32*480*128;

  float* opeB[2]  = { alloc(S_OPE),  alloc(S_OPE)  };
  float* maB[2]   = { alloc(S_MA),   alloc(S_MA)   };
  float* vehB[2]  = { alloc(S_VEH),  alloc(S_VEH)  };
  float* procB[2] = { alloc(S_PROC), alloc(S_PROC) };
  float* onlB[2]  = { alloc(S_ONL),  alloc(S_ONL)  };
  float* offB[2]  = { alloc(S_OFF),  alloc(S_OFF)  };
  float* node_in  = alloc(S_NODE);
  float* node_out = alloc(S_NODE);
  float* xcol     = alloc((size_t)32*450*128);
  float* edge_in  = alloc((size_t)32*480*480);
  float* po       = alloc((size_t)32*400*80);
  float* opb      = alloc((size_t)32*50*450);
  float* eoC      = alloc(S_OFF);
  float* qb  = alloc(S_NODE); float* kb  = alloc(S_NODE); float* vb = alloc(S_NODE);
  float* ob  = alloc(S_NODE); float* pb  = alloc(S_NODE); float* x1 = alloc(S_NODE);
  float* hb  = alloc((size_t)32*480*512);

  // ---- init state from inputs (d2d async copies; graph-capture safe) ----
  hipMemcpyAsync(opeB[0],  d_in[0], S_OPE  * 4, hipMemcpyDeviceToDevice, stream);
  hipMemcpyAsync(maB[0],   d_in[1], S_MA   * 4, hipMemcpyDeviceToDevice, stream);
  hipMemcpyAsync(vehB[0],  d_in[2], S_VEH  * 4, hipMemcpyDeviceToDevice, stream);
  hipMemcpyAsync(procB[0], d_in[3], S_PROC * 4, hipMemcpyDeviceToDevice, stream);
  hipMemcpyAsync(onlB[0],  d_in[5], S_ONL  * 4, hipMemcpyDeviceToDevice, stream);
  hipMemcpyAsync(offB[0],  d_in[6], S_OFF  * 4, hipMemcpyDeviceToDevice, stream);

  int s = 0;
  for (int l = 0; l < 3; ++l) {
    float *ope = opeB[s], *ma = maB[s], *veh = vehB[s];
    float *proc = procB[s], *onl = onlB[s], *ofv = offB[s];
    float *opeN = opeB[1-s], *maN = maB[1-s], *vehN = vehB[1-s];
    float *procN = procB[1-s], *onlN = onlB[1-s], *ofvN = offB[1-s];

    // Block A: rows = ope(400), cols = [ma|veh](80), edge = [proc | off_OV^T]
    { int t = 32*80*128;
      concat2_kernel<<<cdiv(t,256),256,0,stream>>>(xcol, ma, veh, 50, 30); }
    { int t = 32*400*80;
      build_edgeA_kernel<<<cdiv(t,256),256,0,stream>>>(edge_in, proc, ofv); }
    run_block(ope, 400, xcol, 80, edge_in, po, blk_weights(d_in, 3*l),
              opeN, qb, kb, vb, ob, pb, x1, hb, stream);

    // Block B: rows = ma(50), cols = [ma|ope](450), edge = [onload | proc^T]
    { int t = 32*450*128;
      concat2_kernel<<<cdiv(t,256),256,0,stream>>>(xcol, ma, ope, 50, 400); }
    { int t = 32*50*450;
      build_edgeB_kernel<<<cdiv(t,256),256,0,stream>>>(edge_in, onl, proc); }
    run_block(ma, 50, xcol, 450, edge_in, opb, blk_weights(d_in, 3*l+1),
              maN, qb, kb, vb, ob, pb, x1, hb, stream);

    // Block C: rows = veh(30), cols = ope(400), edge = off_OV
    run_block(veh, 30, ope, 400, ofv, eoC, blk_weights(d_in, 3*l+2),
              vehN, qb, kb, vb, ob, pb, x1, hb, stream);

    // Edge-state updates.
    { int t = 32*400*50;
      update_proc_kernel<<<cdiv(t,256),256,0,stream>>>(procN, po, opb); }
    { int t = 32*30*400;
      update_offov_kernel<<<cdiv(t,256),256,0,stream>>>(ofvN, po, eoC); }
    { int t = 32*50*50;
      update_onload_kernel<<<cdiv(t,256),256,0,stream>>>(onlN, opb); }
    s ^= 1;
  }

  // ---- final block over concatenated nodes (480) ----
  { int t = 32*480*128;
    concat3_kernel<<<cdiv(t,256),256,0,stream>>>(node_in, opeB[s], maB[s], vehB[s],
                                                 400, 50, 30); }
  { int t = 32*480*480;
    build_ge_kernel<<<cdiv(t,256),256,0,stream>>>(edge_in, procB[s], onlB[s], offB[s]); }
  run_block(node_in, 480, node_in, 480, edge_in, /*edge_out=*/nullptr,
            blk_weights(d_in, 9), node_out, qb, kb, vb, ob, pb, x1, hb, stream);

  // ---- write outputs: (ope, ma, veh, proc, onload, off_OV) concat-flat ----
  float* out = (float*)d_out;
  size_t o = 0;
  { int t = (int)S_OPE;
    slice_kernel<<<cdiv(t,256),256,0,stream>>>(out + o, node_out, 0, 400, 480); o += S_OPE; }
  { int t = (int)S_MA;
    slice_kernel<<<cdiv(t,256),256,0,stream>>>(out + o, node_out, 400, 50, 480); o += S_MA; }
  { int t = (int)S_VEH;
    slice_kernel<<<cdiv(t,256),256,0,stream>>>(out + o, node_out, 450, 30, 480); o += S_VEH; }
  hipMemcpyAsync(out + o, procB[s], S_PROC * 4, hipMemcpyDeviceToDevice, stream); o += S_PROC;
  hipMemcpyAsync(out + o, onlB[s],  S_ONL  * 4, hipMemcpyDeviceToDevice, stream); o += S_ONL;
  hipMemcpyAsync(out + o, offB[s],  S_OFF  * 4, hipMemcpyDeviceToDevice, stream);
}
